// cross_AFT_77919296684517
// MI455X (gfx1250) — matrix operand order repaired and checked by reading, not compiled, not run on hardware
//
#include <hip/hip_runtime.h>
#include <hip/hip_bf16.h>

// ---------------------------------------------------------------------------
// Cross-AFT for MI455X (gfx1250, wave32, WMMA bf16 16x16x32, f32 accumulate)
// ---------------------------------------------------------------------------
#define BB   64
#define TD   256
#define TP   1024
#define DD   128
#define NEGV (-1e9f)

typedef __attribute__((ext_vector_type(16))) __bf16 v16bf;
typedef __attribute__((ext_vector_type(8)))  float  v8f;

// --- helpers ---------------------------------------------------------------

__device__ __forceinline__ float fast_rcp(float x) {
  return __builtin_amdgcn_rcpf(x);          // v_rcp_f32, no div-fixup chain
}

__device__ __forceinline__ unsigned int pack2bf(float a, float b) {
  unsigned short lo = __builtin_bit_cast(unsigned short, (__bf16)a);
  unsigned short hi = __builtin_bit_cast(unsigned short, (__bf16)b);
  return (unsigned int)lo | ((unsigned int)hi << 16);
}

// A fragment (16x32 bf16): lane = row (l%16), one contiguous 16-half run at
// K offset h*16  (h = l/16).  Two ds_load_b128 from LDS.
__device__ __forceinline__ v16bf load_frag_lds(const __bf16* row, int h) {
  union { v16bf v; uint4 q[2]; } u;
  u.q[0] = *(const uint4*)(row + h * 16);
  u.q[1] = *(const uint4*)(row + h * 16 + 8);
  return u.v;
}

// B fragment pre-packed in global memory: 8 consecutive u32 per lane.
__device__ __forceinline__ v16bf load_frag_glb(const unsigned int* p) {
  union { v16bf v; uint4 q[2]; } u;
  u.q[0] = *(const uint4*)(p);
  u.q[1] = *(const uint4*)(p + 4);
  return u.v;
}

__device__ __forceinline__ v8f wmma_bf16(v16bf a, v16bf b, v8f c) {
  return __builtin_amdgcn_wmma_f32_16x16x32_bf16(
      /*neg_a=*/false, a, /*neg_b=*/false, b,
      /*c_mod=*/(short)0, c, /*reuse_a=*/false, /*reuse_b=*/false);
}

// --- kernel 1: pack a 128x128 weight (Y = X @ W^T => B[k][o] = W[o][k]) into
// bf16 B-fragment layout: [kc(4)][nb(8)][lane(32)][v(8)] u32 ------------------
__global__ void k_pack_w(const float* __restrict__ W,
                         unsigned int* __restrict__ out) {
  int gid  = blockIdx.x * blockDim.x + threadIdx.x;   // 0..8191
  int v    = gid & 7;
  int lane = (gid >> 3) & 31;
  int nb   = (gid >> 8) & 7;
  int kc   = gid >> 11;
  int n = lane & 15, h = lane >> 4;
  int o  = nb * 16 + n;
  int k0 = kc * 32 + h * 16 + 2 * v;
  out[gid] = pack2bf(W[o * DD + k0], W[o * DD + k0 + 1]);
}

// --- kernel 2: projection GEMM  Y[b,t,:] = act(X[b,t,:] @ W^T + bias) -------
#define XS 136   // LDS row stride in halves (272B, 16B aligned, conflict-free)
__global__ void k_proj(const float* __restrict__ X,
                       const unsigned int* __restrict__ wpk,
                       const float* __restrict__ bias,
                       float* __restrict__ Y, int T, int act) {
  __shared__ __align__(16) __bf16 xl[16 * XS];
  int b = blockIdx.y, i0 = blockIdx.x * 16;
  int tid = threadIdx.x;
  {   // stage X tile (16x128 f32 -> bf16 LDS), 8 elements / thread
    int i  = tid >> 4;
    int d0 = (tid & 15) * 8;
    const float* src = X + ((size_t)b * T + i0 + i) * DD + d0;
    float4 f0 = *(const float4*)src;
    float4 f1 = *(const float4*)(src + 4);
    __bf16* dst = xl + i * XS + d0;
    dst[0] = (__bf16)f0.x; dst[1] = (__bf16)f0.y;
    dst[2] = (__bf16)f0.z; dst[3] = (__bf16)f0.w;
    dst[4] = (__bf16)f1.x; dst[5] = (__bf16)f1.y;
    dst[6] = (__bf16)f1.z; dst[7] = (__bf16)f1.w;
  }
  __syncthreads();
  int lane = tid & 31, wv = tid >> 5;
  int n = lane & 15, h = lane >> 4;
  v8f acc = {0.f, 0.f, 0.f, 0.f, 0.f, 0.f, 0.f, 0.f};
  for (int kc = 0; kc < 4; ++kc) {
    v16bf a  = load_frag_lds(xl + n * XS + kc * 32, h);
    v16bf bb = load_frag_glb(wpk + ((kc * 8 + wv) * 32 + lane) * 8);
    acc = wmma_bf16(a, bb, acc);
  }
  int col = wv * 16 + n;
  float bs = bias[col];
  for (int r = 0; r < 8; ++r) {
    int   m   = r + 8 * h;                      // C layout: M = r + 8*(lane/16)
    float val = acc[r] + bs;
    if (act) val = fast_rcp(1.0f + __expf(-val));   // sigmoid via v_rcp_f32
    Y[((size_t)b * T + i0 + m) * DD + col] = val;
  }
}

// --- kernel 3: per-(b,d) max over sequence of raw k -------------------------
__global__ void k_colmax(const float* __restrict__ K,
                         float* __restrict__ mx, int T) {
  int idx = blockIdx.x * blockDim.x + threadIdx.x;  // BB*DD
  int b = idx >> 7, d = idx & 127;
  const float* p = K + (size_t)b * T * DD + d;
  float m = -3.4e38f;
  for (int t = 0; t < T; ++t) m = fmaxf(m, p[(size_t)t * DD]);
  mx[idx] = m;
}

// --- kernel 4: pack (exp(k-max)*v | exp(k-max)) into bf16 B-fragment layout:
// [b][kc(T/32)][ct(16)][lane(32)][v(8)] u32 ---------------------------------
__global__ void k_pack_data(const float* __restrict__ Kr,
                            const float* __restrict__ Vr,
                            const float* __restrict__ mx,
                            unsigned int* __restrict__ pk, int T) {
  int b = blockIdx.y, kc = blockIdx.x;
  int tid  = threadIdx.x;            // 512 threads = 16 ct x 32 lane
  int lane = tid & 31, ct = tid >> 5;
  int n = lane & 15, h = (lane >> 4) & 1;
  int ekv = (ct < 8);
  int c   = (ekv ? ct : ct - 8) * 16 + n;
  float mv = mx[b * DD + c];
  size_t base = ((((size_t)b * (T / 32) + kc) * 16 + ct) * 32 + lane) * 8;
  for (int v = 0; v < 8; ++v) {
    int j0 = kc * 32 + h * 16 + 2 * v;
    float e0 = __expf(Kr[((size_t)b * T + j0) * DD + c] - mv);
    float e1 = __expf(Kr[((size_t)b * T + j0 + 1) * DD + c] - mv);
    if (ekv) {
      e0 *= Vr[((size_t)b * T + j0) * DD + c];
      e1 *= Vr[((size_t)b * T + j0 + 1) * DD + c];
    }
    pk[base + v] = pack2bf(e0, e1);
  }
}

// --- kernel 5: row max of masked position bias ------------------------------
__global__ void k_rowmax(const int* __restrict__ qm, const int* __restrict__ km,
                         const float* __restrict__ pb, float* __restrict__ mpb,
                         int Ti, int Tj, int transposed) {
  int idx = blockIdx.x * blockDim.x + threadIdx.x;  // BB*Ti
  int b = idx / Ti, i = idx % Ti;
  int q = qm[idx];
  float m = NEGV;
  for (int j = 0; j < Tj; ++j) {
    int comb = q * km[b * Tj + j];
    float v = comb ? (transposed ? pb[(size_t)j * 1024 + i]
                                 : pb[(size_t)i * 1024 + j])
                   : NEGV;
    m = fmaxf(m, v);
  }
  mpb[idx] = m;
}

// --- kernel 6: AFT core. Block = (query tile of 16, batch). 8 waves own the
// 16 combined output column tiles (num|den, 256 cols). Double-buffered
// exp_pb staging (one barrier / K-chunk), then fused sigmoid(q)*num/den,
// bf16 output projection, masked 16-row max -> partial. ---------------------
#define AS  40    // exp_pb LDS row stride (halves): 80B, conflict-free b128
#define NDS 264   // num|den LDS row stride (floats)
#define YS  136   // y LDS row stride (halves)
__global__ void k_aft_core(const float* __restrict__ mpb,
                           const float* __restrict__ pb,
                           const int* __restrict__ qm,
                           const int* __restrict__ km,
                           const float* __restrict__ qsig,
                           const unsigned int* __restrict__ packB,
                           const unsigned int* __restrict__ packW,
                           float* __restrict__ partial,
                           int Ti, int Tj, int transposed) {
  __shared__ __align__(16) __bf16 at[2][16 * AS];   // double-buffered
  __shared__ float                nd[16 * NDS];
  __shared__ __align__(16) __bf16 yl[16 * YS];
  __shared__ float mpbs[16];
  __shared__ int   qms[16];

  int b = blockIdx.y, it = blockIdx.x, i0 = it * 16;
  int tid = threadIdx.x;
  if (tid < 16) {
    mpbs[tid] = mpb[b * Ti + i0 + tid];
    qms[tid]  = qm[b * Ti + i0 + tid];
  }
  __syncthreads();

  int lane = tid & 31, wv = tid >> 5;
  int n = lane & 15, h = lane >> 4;
  int nkc = Tj / 32;
  int ct0 = 2 * wv, ct1 = 2 * wv + 1;
  v8f acc0 = {0.f, 0.f, 0.f, 0.f, 0.f, 0.f, 0.f, 0.f};
  v8f acc1 = {0.f, 0.f, 0.f, 0.f, 0.f, 0.f, 0.f, 0.f};

  // stage exp(pb_masked - rowmax) tile (16x32) as bf16 into LDS buffer
  auto fill_tile = [&](int kc, __bf16* dst) {
    for (int e = tid; e < 512; e += 256) {
      int i = e >> 5, jj = e & 31;
      int j = kc * 32 + jj;
      int comb = qms[i] * km[b * Tj + j];
      float pv = comb ? (transposed ? pb[(size_t)j * 1024 + (i0 + i)]
                                    : pb[(size_t)(i0 + i) * 1024 + j])
                      : NEGV;
      dst[i * AS + jj] = (__bf16)__expf(pv - mpbs[i]);
    }
  };

  fill_tile(0, at[0]);
  __syncthreads();

  for (int kc = 0; kc < nkc; ++kc) {
    const __bf16* cur = at[kc & 1];
    __bf16*       nxt = at[(kc + 1) & 1];
    const unsigned int* bp = packB + ((size_t)b * nkc + kc) * 4096;

    v16bf a  = load_frag_lds(cur + n * AS, h);
    v16bf b0 = load_frag_glb(bp + (ct0 * 32 + lane) * 8);
    v16bf b1 = load_frag_glb(bp + (ct1 * 32 + lane) * 8);
    acc0 = wmma_bf16(a, b0, acc0);
    acc1 = wmma_bf16(a, b1, acc1);

    if (kc + 1 < nkc) {
      // prefetch next packed K-chunk (16KB spread across the block)
      __builtin_prefetch(bp + 4096 + tid * 16, 0, 1);
      fill_tile(kc + 1, nxt);   // writes the other buffer: overlaps the WMMAs
    }
    __syncthreads();
  }

  // exchange num/den tiles through LDS
  for (int r = 0; r < 8; ++r) {
    int m = r + 8 * h;
    nd[m * NDS + ct0 * 16 + n] = acc0[r];
    nd[m * NDS + ct1 * 16 + n] = acc1[r];
  }
  __syncthreads();

  // y = sigmoid(q) * num / den  ->  bf16 LDS  (8 elements / thread)
  {
    int i = tid >> 4, d0 = (tid & 15) * 8;
    const float* qp = qsig + ((size_t)b * Ti + i0 + i) * DD + d0;
    for (int k = 0; k < 8; ++k) {
      int d = d0 + k;
      float num = nd[i * NDS + d];
      float den = nd[i * NDS + 128 + d];
      yl[i * YS + d] = (__bf16)(qp[k] * num * fast_rcp(den));
    }
  }
  __syncthreads();

  // output projection: wave wv owns final column tile wv (16 cols of 128)
  v8f acc = {0.f, 0.f, 0.f, 0.f, 0.f, 0.f, 0.f, 0.f};
  for (int kc = 0; kc < 4; ++kc) {
    v16bf a  = load_frag_lds(yl + n * YS + kc * 32, h);
    v16bf bb = load_frag_glb(packW + ((kc * 8 + wv) * 32 + lane) * 8);
    acc = wmma_bf16(a, bb, acc);
  }

  // masked max over this tile's 16 query rows
  float m8 = NEGV;
  for (int r = 0; r < 8; ++r) {
    int m = r + 8 * h;
    m8 = fmaxf(m8, qms[m] ? acc[r] : NEGV);
  }
  float tm = fmaxf(m8, __shfl_xor(m8, 16, 32));
  if (lane < 16)
    partial[((size_t)b * (Ti / 16) + it) * DD + wv * 16 + n] = tm;
}

// --- kernel 7: reduce partial tile maxima, add output bias ------------------
__global__ void k_final(const float* __restrict__ pd,
                        const float* __restrict__ pp,
                        const float* __restrict__ bpd,
                        const float* __restrict__ bpp,
                        float* __restrict__ out) {
  int idx = blockIdx.x * blockDim.x + threadIdx.x;  // 2*BB*DD
  int side = idx >> 13;
  int r = idx & 8191;
  int b = r >> 7, d = r & 127;
  const float* p; int nt; const float* bias;
  if (side == 0) { p = pd + (size_t)b * 16 * DD + d; nt = 16; bias = bpd; }
  else           { p = pp + (size_t)b * 64 * DD + d; nt = 64; bias = bpp; }
  float m = NEGV;
  for (int t = 0; t < nt; ++t) m = fmaxf(m, p[(size_t)t * DD]);
  out[idx] = m + bias[d];
}

// ---------------------------------------------------------------------------
extern "C" void kernel_launch(void* const* d_in, const int* in_sizes, int n_in,
                              void* d_out, int out_size, void* d_ws,
                              size_t ws_size, hipStream_t stream) {
  (void)in_sizes; (void)n_in; (void)out_size; (void)ws_size;
  const float* smiles  = (const float*)d_in[0];
  const float* protein = (const float*)d_in[1];
  const int*   smask   = (const int*)d_in[2];
  const int*   pmask   = (const int*)d_in[3];
  const float* pb      = (const float*)d_in[4];
  const float* W[8] = {(const float*)d_in[5],  (const float*)d_in[7],
                       (const float*)d_in[9],  (const float*)d_in[11],
                       (const float*)d_in[13], (const float*)d_in[15],
                       (const float*)d_in[17], (const float*)d_in[19]};
  const float* bqd = (const float*)d_in[6];
  const float* bkp = (const float*)d_in[8];
  const float* bvp = (const float*)d_in[10];
  const float* bqp = (const float*)d_in[12];
  const float* bkd = (const float*)d_in[14];
  const float* bvd = (const float*)d_in[16];
  const float* bpd = (const float*)d_in[18];
  const float* bpp = (const float*)d_in[20];

  char* base = (char*)d_ws;
  size_t off = 0;
  auto alloc = [&](size_t bytes) -> char* {
    off = (off + 255) & ~(size_t)255;
    char* p = base + off;
    off += bytes;
    return p;
  };
  float* qd_sig = (float*)alloc((size_t)BB * TD * DD * 4);
  float* qp_sig = (float*)alloc((size_t)BB * TP * DD * 4);
  float* kpr    = (float*)alloc((size_t)BB * TP * DD * 4);
  float* vpr    = (float*)alloc((size_t)BB * TP * DD * 4);
  float* kdr    = (float*)alloc((size_t)BB * TD * DD * 4);
  float* vdr    = (float*)alloc((size_t)BB * TD * DD * 4);
  float* mxp    = (float*)alloc((size_t)BB * DD * 4);
  float* mxd    = (float*)alloc((size_t)BB * DD * 4);
  float* mpbd   = (float*)alloc((size_t)BB * TD * 4);
  float* mpbp   = (float*)alloc((size_t)BB * TP * 4);
  unsigned int* packp = (unsigned int*)alloc((size_t)BB * TP * 256 * 2);
  unsigned int* packd = (unsigned int*)alloc((size_t)BB * TD * 256 * 2);
  unsigned int* wpack = (unsigned int*)alloc((size_t)8 * 8192 * 4);
  float* partd  = (float*)alloc((size_t)BB * 16 * DD * 4);
  float* partp  = (float*)alloc((size_t)BB * 64 * DD * 4);

  // 1. pack all weight matrices into bf16 B-fragment layout
  for (int w = 0; w < 8; ++w)
    k_pack_w<<<32, 256, 0, stream>>>(W[w], wpack + w * 8192);

  // 2. input projections (sigmoid fused for queries)
  k_proj<<<dim3(TD / 16, BB), 256, 0, stream>>>(smiles,  wpack + 0 * 8192, bqd, qd_sig, TD, 1);
  k_proj<<<dim3(TP / 16, BB), 256, 0, stream>>>(protein, wpack + 1 * 8192, bkp, kpr,    TP, 0);
  k_proj<<<dim3(TP / 16, BB), 256, 0, stream>>>(protein, wpack + 2 * 8192, bvp, vpr,    TP, 0);
  k_proj<<<dim3(TP / 16, BB), 256, 0, stream>>>(protein, wpack + 3 * 8192, bqp, qp_sig, TP, 1);
  k_proj<<<dim3(TD / 16, BB), 256, 0, stream>>>(smiles,  wpack + 4 * 8192, bkd, kdr,    TD, 0);
  k_proj<<<dim3(TD / 16, BB), 256, 0, stream>>>(smiles,  wpack + 5 * 8192, bvd, vdr,    TD, 0);

  // 3. per-(b,d) key maxima
  k_colmax<<<(BB * DD) / 256, 256, 0, stream>>>(kpr, mxp, TP);
  k_colmax<<<(BB * DD) / 256, 256, 0, stream>>>(kdr, mxd, TD);

  // 4. pack (exp(k)*v | exp(k)) bf16 fragments
  k_pack_data<<<dim3(TP / 32, BB), 512, 0, stream>>>(kpr, vpr, mxp, packp, TP);
  k_pack_data<<<dim3(TD / 32, BB), 512, 0, stream>>>(kdr, vdr, mxd, packd, TD);

  // 5. masked pos-bias row maxima
  k_rowmax<<<(BB * TD) / 256, 256, 0, stream>>>(smask, pmask, pb, mpbd, TD, TP, 0);
  k_rowmax<<<(BB * TP) / 256, 256, 0, stream>>>(pmask, smask, pb, mpbp, TP, TD, 1);

  // 6. AFT cores (drug queries over protein keys, then the reverse)
  k_aft_core<<<dim3(TD / 16, BB), 256, 0, stream>>>(
      mpbd, pb, smask, pmask, qd_sig, packp, wpack + 6 * 8192, partd, TD, TP, 0);
  k_aft_core<<<dim3(TP / 16, BB), 256, 0, stream>>>(
      mpbp, pb, pmask, smask, qp_sig, packd, wpack + 7 * 8192, partp, TP, TD, 1);

  // 7. final masked max + bias
  k_final<<<(2 * BB * DD) / 256, 256, 0, stream>>>(partd, partp, bpd, bpp,
                                                   (float*)d_out);
}
